// RelativeMultiHeadAttenion_83648783057626
// MI455X (gfx1250) — matrix operand order repaired and checked by reading, not compiled, not run on hardware
//
#include <hip/hip_runtime.h>
#include <hip/hip_bf16.h>
#include <stdint.h>

#define B_   2
#define S_   1024
#define D_   1024
#define H_   16
#define DH_  64

typedef __bf16 bf16_t;
typedef __attribute__((ext_vector_type(16))) __bf16 v16bf;
typedef __attribute__((ext_vector_type(8)))  float  v8f;

union FragIn  { v16bf v; unsigned int u[8]; };
union FragAcc { v8f   v; float        f[8]; };

// LDS tiles are [row][34] (bf16), padded to 34 elems (68B, odd dword count) to spread banks.
// A tile stored [m][k]; B tile stored [n][k] (K-transposed) so both fragment loads are b32 dword loads.

__device__ __forceinline__ void load_frag_a(const bf16_t* tile, int mbase, int lane, FragIn& f) {
    const unsigned int* row = (const unsigned int*)(tile + (size_t)(mbase + (lane & 15)) * 34);
    const int kh = (lane >> 4) << 3;             // lanes>=16 hold K+8 block
#pragma unroll
    for (int vv = 0; vv < 8; ++vv) {
        const int k0 = ((vv & 3) << 1) + kh + ((vv >> 2) << 4);  // even
        f.u[vv] = row[k0 >> 1];
    }
}

__device__ __forceinline__ void load_frag_b(const bf16_t* tile, int nbase, int lane, FragIn& f) {
    const unsigned int* row = (const unsigned int*)(tile + (size_t)(nbase + (lane & 15)) * 34);
    const int kw = (lane >> 4) << 2;             // dword offset 4 => K base 8 for upper lanes
#pragma unroll
    for (int vv = 0; vv < 8; ++vv) f.u[vv] = row[kw + (vv & 3) + ((vv >> 2) << 3)];
}

#define WMMA_BF16(acc, fa, fb) \
    (acc).v = __builtin_amdgcn_wmma_f32_16x16x32_bf16(false, (fa).v, false, (fb).v, (short)0, (acc).v, false, false)

// ---------------------------------------------------------------------------
// Big-K GEMM: C[2048x1024] = A[2048x1024] x W[1024x1024]
// OUT_MODE 0: bf16 out in head-split layout [B,H,S,DH];  OUT_MODE 1: f32 row-major.
// Block: 128 threads = 4 waves; block tile 128x128; each wave 64x64 (4x4 WMMA frags); K-step 32.
// ---------------------------------------------------------------------------
template<bool A_BF16, int OUT_MODE>
__global__ __launch_bounds__(128)
void gemm_k1024(const void* __restrict__ Ap, const float* __restrict__ W, void* __restrict__ outp)
{
    constexpr int K = 1024, N = 1024;
    __shared__ bf16_t lds_a[128 * 34];
    __shared__ bf16_t lds_b[128 * 34];
    const int tid = threadIdx.x, lane = tid & 31, wave = tid >> 5;
    const int bm = blockIdx.y * 128, bn = blockIdx.x * 128;
    const int wm = (wave >> 1) * 64, wn = (wave & 1) * 64;

    FragAcc acc[4][4];
#pragma unroll
    for (int i = 0; i < 4; ++i)
#pragma unroll
        for (int j = 0; j < 4; ++j)
#pragma unroll
            for (int e = 0; e < 8; ++e) acc[i][j].f[e] = 0.0f;

    for (int k0 = 0; k0 < K; k0 += 32) {
        if (A_BF16) {
            const unsigned int* src = (const unsigned int*)((const bf16_t*)Ap + (size_t)(bm + tid) * K + k0);
            unsigned int* dst = (unsigned int*)(lds_a + (size_t)tid * 34);
#pragma unroll
            for (int i = 0; i < 8; ++i) dst[i] = src[i];
        } else {
            const float* src = (const float*)Ap + (size_t)(bm + tid) * K + k0;
            bf16_t* dst = lds_a + (size_t)tid * 34;
#pragma unroll
            for (int i = 0; i < 32; ++i) dst[i] = (bf16_t)src[i];
        }
        {   // W[k][n] -> lds_b[n][k]; coalesced across threads per k
            bf16_t* dst = lds_b + (size_t)tid * 34;
            const float* src = W + (size_t)k0 * N + bn + tid;
#pragma unroll 4
            for (int kk = 0; kk < 32; ++kk) dst[kk] = (bf16_t)src[(size_t)kk * N];
        }
        __syncthreads();
        FragIn fa[4], fb[4];
#pragma unroll
        for (int i = 0; i < 4; ++i) load_frag_a(lds_a, wm + 16 * i, lane, fa[i]);
#pragma unroll
        for (int j = 0; j < 4; ++j) load_frag_b(lds_b, wn + 16 * j, lane, fb[j]);
#pragma unroll
        for (int i = 0; i < 4; ++i)
#pragma unroll
            for (int j = 0; j < 4; ++j) WMMA_BF16(acc[i][j], fa[i], fb[j]);
        __syncthreads();
    }

    const int mo = (lane >> 4) << 3, nlane = lane & 15;
#pragma unroll
    for (int i = 0; i < 4; ++i)
#pragma unroll
        for (int j = 0; j < 4; ++j) {
            const int n = bn + wn + 16 * j + nlane;
#pragma unroll
            for (int v = 0; v < 8; ++v) {
                const int m = bm + wm + 16 * i + mo + v;
                const float val = acc[i][j].f[v];
                if (OUT_MODE == 0) {
                    const int b = m >> 10, s = m & 1023, h = n >> 6, dd = n & 63;
                    ((bf16_t*)outp)[((size_t)((b * H_ + h) * S_ + s)) * DH_ + dd] = (bf16_t)val;
                } else {
                    ((float*)outp)[(size_t)m * N + n] = val;
                }
            }
        }
}

// ---------------------------------------------------------------------------
// Per-head (S x DH) x (DH x S) GEMM: A = q_bh, B = rows^T (Er or k_bh).
// MODE 0: write QEr (f32).  MODE 1: fused skew-gather + scale -> scores (f32).
// ---------------------------------------------------------------------------
template<bool B_BF16, int MODE>
__global__ __launch_bounds__(128)
void attn_scores(const bf16_t* __restrict__ qall, const void* __restrict__ Brows,
                 float* __restrict__ QErB, float* __restrict__ SCB)
{
    __shared__ bf16_t lds_a[128 * 34];
    __shared__ bf16_t lds_b[128 * 34];
    const int tid = threadIdx.x, lane = tid & 31, wave = tid >> 5;
    const int bh = blockIdx.z;
    const int bm = blockIdx.y * 128, bn = blockIdx.x * 128;
    const int wm = (wave >> 1) * 64, wn = (wave & 1) * 64;
    const bf16_t* A = qall + (size_t)bh * S_ * DH_;

    FragAcc acc[4][4];
#pragma unroll
    for (int i = 0; i < 4; ++i)
#pragma unroll
        for (int j = 0; j < 4; ++j)
#pragma unroll
            for (int e = 0; e < 8; ++e) acc[i][j].f[e] = 0.0f;

#pragma unroll
    for (int k0 = 0; k0 < DH_; k0 += 32) {
        {
            const unsigned int* src = (const unsigned int*)(A + (size_t)(bm + tid) * DH_ + k0);
            unsigned int* dst = (unsigned int*)(lds_a + (size_t)tid * 34);
#pragma unroll
            for (int i = 0; i < 8; ++i) dst[i] = src[i];
        }
        if (B_BF16) {
            const bf16_t* Bm = (const bf16_t*)Brows + (size_t)bh * S_ * DH_;
            const unsigned int* src = (const unsigned int*)(Bm + (size_t)(bn + tid) * DH_ + k0);
            unsigned int* dst = (unsigned int*)(lds_b + (size_t)tid * 34);
#pragma unroll
            for (int i = 0; i < 8; ++i) dst[i] = src[i];
        } else {
            const float* src = (const float*)Brows + (size_t)(bn + tid) * DH_ + k0;
            bf16_t* dst = lds_b + (size_t)tid * 34;
#pragma unroll
            for (int i = 0; i < 32; ++i) dst[i] = (bf16_t)src[i];
        }
        __syncthreads();
        FragIn fa[4], fb[4];
#pragma unroll
        for (int i = 0; i < 4; ++i) load_frag_a(lds_a, wm + 16 * i, lane, fa[i]);
#pragma unroll
        for (int j = 0; j < 4; ++j) load_frag_b(lds_b, wn + 16 * j, lane, fb[j]);
#pragma unroll
        for (int i = 0; i < 4; ++i)
#pragma unroll
            for (int j = 0; j < 4; ++j) WMMA_BF16(acc[i][j], fa[i], fb[j]);
        __syncthreads();
    }

    const int mo = (lane >> 4) << 3, nlane = lane & 15;
    const size_t base = (size_t)bh * S_ * S_;
#pragma unroll
    for (int i = 0; i < 4; ++i)
#pragma unroll
        for (int j = 0; j < 4; ++j) {
            const int n = bn + wn + 16 * j + nlane;
#pragma unroll
            for (int v = 0; v < 8; ++v) {
                const int m = bm + wm + 16 * i + mo + v;
                const float val = acc[i][j].f[v];
                if (MODE == 0) {
                    QErB[base + (size_t)m * S_ + n] = val;
                } else {
                    // Srel via exact _skew semantics (incl. unmasked spill region)
                    float srel;
                    if (n <= m)          srel = QErB[base + (size_t)m * S_ + (S_ - 1 - m + n)];
                    else if (n == m + 1) srel = 0.0f;
                    else                 srel = QErB[base + (size_t)(m + 1) * S_ + (n - m - 2)];
                    SCB[base + (size_t)m * S_ + n] = (val + srel) * 0.125f;  // 1/sqrt(64)
                }
            }
        }
}

// ---------------------------------------------------------------------------
// Row softmax over 1024 elems, in place (f32). One block of 128 per row.
// ---------------------------------------------------------------------------
__global__ __launch_bounds__(128)
void softmax_rows(float* __restrict__ SCB)
{
    __shared__ float red[128];
    const int t = threadIdx.x;
    float* p = SCB + (size_t)blockIdx.x * S_;
    float x[8];
    float vmax = -3.0e38f;
#pragma unroll
    for (int e = 0; e < 8; ++e) { x[e] = p[t + 128 * e]; vmax = fmaxf(vmax, x[e]); }
    red[t] = vmax; __syncthreads();
    for (int s = 64; s > 0; s >>= 1) { if (t < s) red[t] = fmaxf(red[t], red[t + s]); __syncthreads(); }
    vmax = red[0]; __syncthreads();
    float sum = 0.0f;
#pragma unroll
    for (int e = 0; e < 8; ++e) { x[e] = __expf(x[e] - vmax); sum += x[e]; }
    red[t] = sum; __syncthreads();
    for (int s = 64; s > 0; s >>= 1) { if (t < s) red[t] += red[t + s]; __syncthreads(); }
    const float inv = 1.0f / red[0];
#pragma unroll
    for (int e = 0; e < 8; ++e) p[t + 128 * e] = x[e] * inv;
}

// ---------------------------------------------------------------------------
// attn(1024x1024, f32->bf16 on staging) x v_bh(1024x64 bf16) -> attnout [B,S,D] bf16.
// Block: 4 waves stacked in M (block tile 256x64), each wave 64x64.
// ---------------------------------------------------------------------------
__global__ __launch_bounds__(128)
void attn_av(const float* __restrict__ P, const bf16_t* __restrict__ vall, bf16_t* __restrict__ attnout)
{
    __shared__ bf16_t lds_a[256 * 34];
    __shared__ bf16_t lds_b[64 * 34];
    const int tid = threadIdx.x, lane = tid & 31, wave = tid >> 5;
    const int bh = blockIdx.y;
    const int b = bh >> 4, h = bh & 15;
    const int bm = blockIdx.x * 256;
    const float*  Pbh = P    + (size_t)bh * S_ * S_;
    const bf16_t* vbh = vall + (size_t)bh * S_ * DH_;

    FragAcc acc[4][4];
#pragma unroll
    for (int i = 0; i < 4; ++i)
#pragma unroll
        for (int j = 0; j < 4; ++j)
#pragma unroll
            for (int e = 0; e < 8; ++e) acc[i][j].f[e] = 0.0f;

    for (int k0 = 0; k0 < S_; k0 += 32) {
#pragma unroll
        for (int rr = 0; rr < 2; ++rr) {
            const int r = tid + rr * 128;
            const float* src = Pbh + (size_t)(bm + r) * S_ + k0;
            bf16_t* dst = lds_a + (size_t)r * 34;
#pragma unroll
            for (int i = 0; i < 32; ++i) dst[i] = (bf16_t)src[i];
        }
        {   // v[k][n] -> lds_b[n][k]; thread covers one (k, 16-wide n chunk)
            const int k = tid >> 2;
            const int nb = (tid & 3) * 16;
            const bf16_t* src = vbh + (size_t)(k0 + k) * DH_ + nb;
#pragma unroll
            for (int i = 0; i < 16; ++i) lds_b[(size_t)(nb + i) * 34 + k] = src[i];
        }
        __syncthreads();
        FragIn fa[4], fb[4];
#pragma unroll
        for (int i = 0; i < 4; ++i) load_frag_a(lds_a, wave * 64 + 16 * i, lane, fa[i]);
#pragma unroll
        for (int j = 0; j < 4; ++j) load_frag_b(lds_b, 16 * j, lane, fb[j]);
#pragma unroll
        for (int i = 0; i < 4; ++i)
#pragma unroll
            for (int j = 0; j < 4; ++j) WMMA_BF16(acc[i][j], fa[i], fb[j]);
        __syncthreads();
    }

    const int mo = (lane >> 4) << 3, nlane = lane & 15;
#pragma unroll
    for (int i = 0; i < 4; ++i)
#pragma unroll
        for (int j = 0; j < 4; ++j) {
            const int dd = 16 * j + nlane;
#pragma unroll
            for (int v = 0; v < 8; ++v) {
                const int s = bm + wave * 64 + 16 * i + mo + v;
                attnout[((size_t)(b * S_ + s)) * D_ + h * DH_ + dd] = (bf16_t)acc[i][j].f[v];
            }
        }
}

// ---------------------------------------------------------------------------
extern "C" void kernel_launch(void* const* d_in, const int* in_sizes, int n_in,
                              void* d_out, int out_size, void* d_ws, size_t ws_size,
                              hipStream_t stream)
{
    const float* query = (const float*)d_in[0];
    const float* key   = (const float*)d_in[1];
    const float* value = (const float*)d_in[2];
    const float* WQ    = (const float*)d_in[3];
    const float* WK    = (const float*)d_in[4];
    const float* WV    = (const float*)d_in[5];
    const float* WM    = (const float*)d_in[6];
    const float* Er    = (const float*)d_in[7];

    const size_t qkvE = (size_t)B_ * H_ * S_ * DH_;   // 2,097,152
    const size_t scE  = (size_t)B_ * H_ * S_ * S_;    // 33,554,432

    char* w = (char*)d_ws;
    bf16_t* qb = (bf16_t*)w;      w += qkvE * sizeof(bf16_t);
    bf16_t* kb = (bf16_t*)w;      w += qkvE * sizeof(bf16_t);
    bf16_t* vb = (bf16_t*)w;      w += qkvE * sizeof(bf16_t);
    bf16_t* ao = (bf16_t*)w;      w += (size_t)B_ * S_ * D_ * sizeof(bf16_t);
    w = (char*)(((uintptr_t)w + 255) & ~(uintptr_t)255);
    float* QErB = (float*)w;      w += scE * sizeof(float);
    float* SCB  = (float*)w;      w += scE * sizeof(float);
    (void)ws_size; (void)n_in; (void)out_size;

    const int maxlen = in_sizes[7] / DH_;
    const float* ErUsed = Er + (size_t)(maxlen - S_) * DH_;

    dim3 blk(128);
    // Projections: q,k,v = x @ W  (bf16 out, head-split layout)
    gemm_k1024<false, 0><<<dim3(8, 16), blk, 0, stream>>>((const void*)query, WQ, (void*)qb);
    gemm_k1024<false, 0><<<dim3(8, 16), blk, 0, stream>>>((const void*)key,   WK, (void*)kb);
    gemm_k1024<false, 0><<<dim3(8, 16), blk, 0, stream>>>((const void*)value, WV, (void*)vb);
    // QEr = q @ Er^T (per head)
    attn_scores<false, 0><<<dim3(8, 8, B_ * H_), blk, 0, stream>>>(qb, (const void*)ErUsed, QErB, SCB);
    // scores = (q @ k^T + skew(QEr)) / sqrt(d)
    attn_scores<true, 1><<<dim3(8, 8, B_ * H_), blk, 0, stream>>>(qb, (const void*)kb, QErB, SCB);
    // softmax rows (in place)
    softmax_rows<<<dim3(B_ * H_ * S_), blk, 0, stream>>>(SCB);
    // out_h = attn @ v, re-interleaved to [B,S,D] bf16
    attn_av<<<dim3(4, B_ * H_), blk, 0, stream>>>(SCB, vb, ao);
    // final = out @ WMerge -> f32 d_out
    gemm_k1024<true, 1><<<dim3(8, 16), blk, 0, stream>>>((const void*)ao, WM, d_out);
}
